// Model_84576495993022
// MI455X (gfx1250) — compile-verified, matrix operand order checked by reading, not run on hardware
//
#include <hip/hip_runtime.h>

typedef __attribute__((ext_vector_type(16))) _Float16 v16h;
typedef __attribute__((ext_vector_type(8)))  float    v8f;
typedef _Float16 f16;

// ---------------------------------------------------------------------------
// Channel permutation within a 32-channel group so that the WMMA A-fragment
// (16-bit A 16x32: lanehi=0 -> K {0..7,16..23}, lanehi=1 -> K {8..15,24..31})
// becomes ONE contiguous 32-byte load per lane.
// position p = 16*lanehi + e  ->  channel c = 8*lanehi + e + (e>=8 ? 8 : 0)
// ---------------------------------------------------------------------------
__device__ __forceinline__ int perm_chan(int p32) {
    int hi = p32 >> 4, e = p32 & 15;
    return hi * 8 + e + (e >= 8 ? 8 : 0);
}

// ---------------------------------------------------------------------------
// Generic gather-GEMM via WMMA:  out[r,:] (+)= sum_k  feat[nbr[r,k],:] @ W_k
// feat  : [(zrow+1) x CIN] f16, channel-permuted, row `zrow` is all zeros
// nbr   : [nOut x KOFF] int32, -1 = inactive
// wpack : [KOFF][CIN/32][COUT/16][32 lanes][16 halves]  (B-fragment layout)
// out   : f32, row stride ldOut, column offset col0
// One wave handles 16 rows x COUT; 8 waves / block.
// ---------------------------------------------------------------------------
template <int CIN, int COUT, int KOFF>
__global__ void __launch_bounds__(256)
subm_conv_wmma(const f16* __restrict__ feat, const int* __restrict__ nbr,
               const f16* __restrict__ wpack, float* __restrict__ out,
               int ldOut, int col0, int nOut, int zrow)
{
    constexpr int KT = CIN / 32;   // K tiles of 32
    constexpr int NT = COUT / 16;  // N tiles of 16
    const int wave   = threadIdx.x >> 5;
    const int lane   = threadIdx.x & 31;
    const int lanelo = lane & 15;
    const int lanehi = lane >> 4;
    const int base   = (blockIdx.x * 8 + wave) * 16;
    if (base >= nOut) return;      // wave-uniform exit

    const int r = base + lanelo;   // row this lane gathers for (A fragment)

    v8f acc[NT] = {};

#pragma unroll 1
    for (int k = 0; k < KOFF; ++k) {
        int j = (r < nOut) ? nbr[(size_t)r * KOFF + k] : -1;
        int jr = (j >= 0) ? j : zrow;          // branch-free: pad row is zeros
#pragma unroll
        for (int kt = 0; kt < KT; ++kt) {
            v16h a = *(const v16h*)(feat + (size_t)jr * CIN + kt * 32 + lanehi * 16);
            const f16* wb = wpack + ((size_t)(k * KT + kt) * NT) * 512 + lane * 16;
#pragma unroll
            for (int nt = 0; nt < NT; ++nt) {
                v16h b = *(const v16h*)(wb + (size_t)nt * 512);
                acc[nt] = __builtin_amdgcn_wmma_f32_16x16x32_f16(
                    false, a, false, b, (short)0, acc[nt], false, false);
            }
        }
    }
    // D layout: VGPR v, lane -> row = base + 8*lanehi + v, col = nt*16 + lanelo
#pragma unroll
    for (int nt = 0; nt < NT; ++nt) {
#pragma unroll
        for (int v = 0; v < 8; ++v) {
            int row = base + lanehi * 8 + v;
            if (row < nOut)
                out[(size_t)row * ldOut + col0 + nt * 16 + lanelo] = acc[nt][v];
        }
    }
}

// ---------------------------------------------------------------------------
// Pack f32 weights [KOFF][CIN][COUT] into the WMMA B-fragment f16 layout:
// wpack[((k*KT+kt)*NT+nt)*512 + L*16 + e] = W[k][kt*32 + perm(16*(L>=16)+e)][nt*16 + (L&15)]
// ---------------------------------------------------------------------------
__global__ void pack_weights(const float* __restrict__ W, f16* __restrict__ wpack,
                             int KOFF, int CIN, int COUT)
{
    int total = KOFF * CIN * COUT;
    for (int idx = blockIdx.x * blockDim.x + threadIdx.x; idx < total;
         idx += gridDim.x * blockDim.x) {
        int e  = idx & 15;
        int L  = (idx >> 4) & 31;
        int t  = idx >> 9;
        int NTl = COUT >> 4, KTl = CIN >> 5;
        int nt = t % NTl;  t /= NTl;
        int kt = t % KTl;
        int k  = t / KTl;
        int p  = ((L >> 4) << 4) + e;                 // K position in 32-group
        int cin  = kt * 32 + perm_chan(p);
        int cout = nt * 16 + (L & 15);
        wpack[idx] = (f16)W[((size_t)k * CIN + cin) * COUT + cout];
    }
}

// ---------------------------------------------------------------------------
// BatchNorm statistics: deterministic two-stage reduction (no atomics).
// ---------------------------------------------------------------------------
template <int C>
__global__ void __launch_bounds__(256)
bn_partial(const float* __restrict__ src, int ld, int col0, int n,
           float* __restrict__ partial /* [grid][C][2] */)
{
    constexpr int RPB = 256 / C;
    const int c = threadIdx.x % C;
    const int rsub = threadIdx.x / C;
    float s = 0.f, ss = 0.f;
    for (int r = blockIdx.x * RPB + rsub; r < n; r += gridDim.x * RPB) {
        float v = src[(size_t)r * ld + col0 + c];
        s += v; ss += v * v;
    }
    __shared__ float ls[256], lss[256];
    ls[threadIdx.x] = s; lss[threadIdx.x] = ss;
    __syncthreads();
    if (threadIdx.x < C) {
        float a = 0.f, b = 0.f;
        for (int i = 0; i < RPB; ++i) { a += ls[i * C + c]; b += lss[i * C + c]; }
        partial[((size_t)blockIdx.x * C + c) * 2 + 0] = a;
        partial[((size_t)blockIdx.x * C + c) * 2 + 1] = b;
    }
}

__global__ void bn_finalize(const float* __restrict__ partial, int nblocks, int C, int n,
                            const float* __restrict__ gamma, const float* __restrict__ beta,
                            float* __restrict__ stats /* [C]{scale,shift} */)
{
    int c = threadIdx.x;
    if (c >= C) return;
    float s = 0.f, ss = 0.f;
    for (int b = 0; b < nblocks; ++b) {
        s  += partial[((size_t)b * C + c) * 2 + 0];
        ss += partial[((size_t)b * C + c) * 2 + 1];
    }
    float mu  = s / (float)n;
    float var = ss / (float)n - mu * mu;
    float sc  = gamma[c] * rsqrtf(var + 1e-4f);
    stats[c * 2 + 0] = sc;
    stats[c * 2 + 1] = beta[c] - mu * sc;
}

// Apply BN+ReLU, emit f16 in permuted channel order; row n (pad) is zeroed.
template <int C>
__global__ void bn_apply_f16(const float* __restrict__ src, int ld, int col0, int n,
                             const float* __restrict__ stats, f16* __restrict__ dst)
{
    int total = (n + 1) * C;
    for (int idx = blockIdx.x * blockDim.x + threadIdx.x; idx < total;
         idx += gridDim.x * blockDim.x) {
        int p = idx % C;
        int r = idx / C;
        int c = (p & ~31) + perm_chan(p & 31);
        float v = 0.f;
        if (r < n) {
            v = src[(size_t)r * ld + col0 + c] * stats[c * 2] + stats[c * 2 + 1];
            v = v > 0.f ? v : 0.f;
        }
        dst[(size_t)r * C + p] = (f16)v;
    }
}

// ---------------------------------------------------------------------------
// Input conv: Cin=3 is too small for WMMA; scalar with weights in LDS.
// ---------------------------------------------------------------------------
__global__ void __launch_bounds__(256)
conv_in_kernel(const float* __restrict__ feats, const int* __restrict__ nbr,
               const float* __restrict__ Win, float* __restrict__ out, int n)
{
    __shared__ float W[27 * 3 * 32];
    for (int i = threadIdx.x; i < 27 * 3 * 32; i += blockDim.x) W[i] = Win[i];
    __syncthreads();
    int nidx = blockIdx.x * blockDim.x + threadIdx.x;
    if (nidx >= n) return;
    float acc[32];
#pragma unroll
    for (int co = 0; co < 32; ++co) acc[co] = 0.f;
    for (int k = 0; k < 27; ++k) {
        int j = nbr[(size_t)nidx * 27 + k];
        if (j < 0) continue;
        float f0 = feats[(size_t)j * 3 + 0];
        float f1 = feats[(size_t)j * 3 + 1];
        float f2 = feats[(size_t)j * 3 + 2];
        const float* wk = W + k * 96;
#pragma unroll
        for (int co = 0; co < 32; ++co)
            acc[co] += f0 * wk[co] + f1 * wk[32 + co] + f2 * wk[64 + co];
    }
#pragma unroll
    for (int co = 0; co < 32; ++co) out[(size_t)nidx * 32 + co] = acc[co];
}

// ---------------------------------------------------------------------------
// Rulebooks for strided down / up conv (deterministic, collision-free).
// ---------------------------------------------------------------------------
__global__ void fill_i32(int* p, int val, int count)
{
    int i = blockIdx.x * blockDim.x + threadIdx.x;
    if (i < count) p[i] = val;
}
__global__ void build_child(const int* __restrict__ parent, const int* __restrict__ offid,
                            int* __restrict__ child, int n)
{
    int i = blockIdx.x * blockDim.x + threadIdx.x;
    if (i < n) child[(size_t)parent[i] * 8 + offid[i]] = i;  // (parent,off) unique
}
__global__ void build_upnbr(const int* __restrict__ parent, const int* __restrict__ offid,
                            int* __restrict__ upnbr, int n)
{
    int i = blockIdx.x * blockDim.x + threadIdx.x;
    if (i < n * 8) {
        int nidx = i >> 3, o = i & 7;
        upnbr[i] = (offid[nidx] == o) ? parent[nidx] : -1;
    }
}

// ---------------------------------------------------------------------------
// Fused top BN+ReLU + linear head (32 -> 50).
// ---------------------------------------------------------------------------
__global__ void __launch_bounds__(256)
bn_linear_kernel(const float* __restrict__ f3, const float* __restrict__ stats,
                 const float* __restrict__ wlin, const float* __restrict__ blin,
                 float* __restrict__ out, int n)
{
    __shared__ float W[32 * 50];
    __shared__ float S[64];
    __shared__ float Bv[50];
    for (int i = threadIdx.x; i < 32 * 50; i += blockDim.x) W[i] = wlin[i];
    for (int i = threadIdx.x; i < 64; i += blockDim.x) S[i] = stats[i];
    for (int i = threadIdx.x; i < 50; i += blockDim.x) Bv[i] = blin[i];
    __syncthreads();
    int idx = blockIdx.x * blockDim.x + threadIdx.x;
    if (idx >= n * 50) return;
    int r = idx / 50, c = idx % 50;
    float acc = Bv[c];
    const float* fr = f3 + (size_t)r * 32;
#pragma unroll
    for (int k = 0; k < 32; ++k) {
        float v = fr[k] * S[k * 2] + S[k * 2 + 1];
        v = v > 0.f ? v : 0.f;
        acc += v * W[k * 50 + c];
    }
    out[idx] = acc;
}

// ---------------------------------------------------------------------------
extern "C" void kernel_launch(void* const* d_in, const int* in_sizes, int n_in,
                              void* d_out, int out_size, void* d_ws, size_t ws_size,
                              hipStream_t stream)
{
    (void)n_in; (void)out_size; (void)ws_size;
    const float* features = (const float*)d_in[1];
    const int*   nbr_fine = (const int*)d_in[2];
    const int*   nbr_crs  = (const int*)d_in[3];
    const int*   parent   = (const int*)d_in[4];
    const int*   offid    = (const int*)d_in[5];
    const float* w_in  = (const float*)d_in[7];
    const float* w_b1  = (const float*)d_in[8];
    const float* w_dn  = (const float*)d_in[9];
    const float* w_b2  = (const float*)d_in[10];
    const float* w_up  = (const float*)d_in[11];
    const float* w_b3  = (const float*)d_in[12];
    const float* g1 = (const float*)d_in[13]; const float* b1 = (const float*)d_in[14];
    const float* gd = (const float*)d_in[15]; const float* bd = (const float*)d_in[16];
    const float* g2 = (const float*)d_in[17]; const float* b2 = (const float*)d_in[18];
    const float* gu = (const float*)d_in[19]; const float* bu = (const float*)d_in[20];
    const float* g3 = (const float*)d_in[21]; const float* b3 = (const float*)d_in[22];
    const float* go = (const float*)d_in[23]; const float* bo = (const float*)d_in[24];
    const float* wlin = (const float*)d_in[25];
    const float* blin = (const float*)d_in[26];

    const int N  = in_sizes[2] / 27;   // fine sites
    const int NC = in_sizes[3] / 27;   // coarse sites

    // ---- workspace carve-up (256-byte aligned) ----
    size_t off = 0;
    auto alloc = [&](size_t bytes) -> void* {
        off = (off + 255) & ~(size_t)255;
        void* p = (char*)d_ws + off;
        off += bytes;
        return p;
    };
    float* A     = (float*)alloc((size_t)N * 64 * 4);        // f1 / g0 / g1c / f3
    float* CAT   = (float*)alloc((size_t)N * 64 * 4);        // [skip | up]
    f16*   H     = (f16*)  alloc((size_t)(N + 1) * 64 * 2);  // f16 perm staging (+pad row)
    int*   child = (int*)  alloc((size_t)NC * 8 * 4);
    int*   upnbr = (int*)  alloc((size_t)N * 8 * 4);
    f16* wp_b1 = (f16*)alloc(27 * 32 * 32 * 2);
    f16* wp_dn = (f16*)alloc( 8 * 32 * 64 * 2);
    f16* wp_b2 = (f16*)alloc(27 * 64 * 64 * 2);
    f16* wp_up = (f16*)alloc( 8 * 64 * 32 * 2);
    f16* wp_b3 = (f16*)alloc(27 * 64 * 32 * 2);
    const int NB = 256;                                      // reduction blocks
    float* part  = (float*)alloc((size_t)NB * 64 * 2 * 4);
    float* stats = (float*)alloc(64 * 2 * 4);

    auto cdiv = [](int a, int b) { return (a + b - 1) / b; };
    auto convGrid = [&](int rows) { return cdiv(cdiv(rows, 16), 8); };

    // ---- pack all conv weights into WMMA B-fragment layout ----
    pack_weights<<<cdiv(27*32*32,256),256,0,stream>>>(w_b1, wp_b1, 27, 32, 32);
    pack_weights<<<cdiv( 8*32*64,256),256,0,stream>>>(w_dn, wp_dn,  8, 32, 64);
    pack_weights<<<cdiv(27*64*64,256),256,0,stream>>>(w_b2, wp_b2, 27, 64, 64);
    pack_weights<<<cdiv( 8*64*32,256),256,0,stream>>>(w_up, wp_up,  8, 64, 32);
    pack_weights<<<cdiv(27*64*32,256),256,0,stream>>>(w_b3, wp_b3, 27, 64, 32);

    // ---- rulebooks for strided down / up conv ----
    fill_i32<<<cdiv(NC*8,256),256,0,stream>>>(child, -1, NC * 8);
    build_child<<<cdiv(N,256),256,0,stream>>>(parent, offid, child, N);
    build_upnbr<<<cdiv(N*8,256),256,0,stream>>>(parent, offid, upnbr, N);

    // 1) input conv (3 -> 32) -> A[N,32]
    conv_in_kernel<<<cdiv(N,256),256,0,stream>>>(features, nbr_fine, w_in, A, N);

    // 2) BN(g1,b1)+ReLU -> H ; conv_b1 (32->32) -> CAT[:, 0:32] (= skip)
    bn_partial<32><<<NB,256,0,stream>>>(A, 32, 0, N, part);
    bn_finalize<<<1,64,0,stream>>>(part, NB, 32, N, g1, b1, stats);
    bn_apply_f16<32><<<cdiv((N+1)*32,256),256,0,stream>>>(A, 32, 0, N, stats, H);
    subm_conv_wmma<32,32,27><<<convGrid(N),256,0,stream>>>(H, nbr_fine, wp_b1, CAT, 64, 0, N, N);

    // 3) BN(gd,bd)+ReLU on skip -> H ; strided down conv (32->64) via child table -> A[NC,64]
    bn_partial<32><<<NB,256,0,stream>>>(CAT, 64, 0, N, part);
    bn_finalize<<<1,64,0,stream>>>(part, NB, 32, N, gd, bd, stats);
    bn_apply_f16<32><<<cdiv((N+1)*32,256),256,0,stream>>>(CAT, 64, 0, N, stats, H);
    subm_conv_wmma<32,64,8><<<convGrid(NC),256,0,stream>>>(H, child, wp_dn, A, 64, 0, NC, N);

    // 4) BN(g2,b2)+ReLU -> H ; conv_b2 (64->64) on coarse -> A[NC,64]
    bn_partial<64><<<NB,256,0,stream>>>(A, 64, 0, NC, part);
    bn_finalize<<<1,64,0,stream>>>(part, NB, 64, NC, g2, b2, stats);
    bn_apply_f16<64><<<cdiv((NC+1)*64,256),256,0,stream>>>(A, 64, 0, NC, stats, H);
    subm_conv_wmma<64,64,27><<<convGrid(NC),256,0,stream>>>(H, nbr_crs, wp_b2, A, 64, 0, NC, NC);

    // 5) BN(gu,bu)+ReLU -> H ; up deconv (64->32) via upnbr -> CAT[:, 32:64]
    bn_partial<64><<<NB,256,0,stream>>>(A, 64, 0, NC, part);
    bn_finalize<<<1,64,0,stream>>>(part, NB, 64, NC, gu, bu, stats);
    bn_apply_f16<64><<<cdiv((NC+1)*64,256),256,0,stream>>>(A, 64, 0, NC, stats, H);
    subm_conv_wmma<64,32,8><<<convGrid(N),256,0,stream>>>(H, upnbr, wp_up, CAT, 64, 32, N, NC);

    // 6) BN(g3,b3)+ReLU on cat[N,64] -> H ; conv_b3 (64->32) -> A[N,32]
    bn_partial<64><<<NB,256,0,stream>>>(CAT, 64, 0, N, part);
    bn_finalize<<<1,64,0,stream>>>(part, NB, 64, N, g3, b3, stats);
    bn_apply_f16<64><<<cdiv((N+1)*64,256),256,0,stream>>>(CAT, 64, 0, N, stats, H);
    subm_conv_wmma<64,32,27><<<convGrid(N),256,0,stream>>>(H, nbr_fine, wp_b3, A, 32, 0, N, N);

    // 7) top BN(go,bo)+ReLU fused with linear head -> d_out[N,50]
    bn_partial<32><<<NB,256,0,stream>>>(A, 32, 0, N, part);
    bn_finalize<<<1,64,0,stream>>>(part, NB, 32, N, go, bo, stats);
    bn_linear_kernel<<<cdiv(N*50,256),256,0,stream>>>(A, stats, wlin, blin, (float*)d_out, N);
}
